// AttentionUNet_27195732918772
// MI455X (gfx1250) — matrix operand order repaired and checked by reading, not compiled, not run on hardware
//
#include <hip/hip_runtime.h>

// ---------------------------------------------------------------------------
// CDNA5 (gfx1250) Attention U-Net forward.
// All matmuls (3x3 convs as implicit GEMM, 1x1 projections, attention QK^T and
// PV) use v_wmma_f32_16x16x32_bf16. Activations are NHWC bf16 so WMMA A/B
// fragment dwords are contiguous-pair loads. Attention is flash-style (online
// softmax) so no N x N score matrix is ever materialized.
// ---------------------------------------------------------------------------

typedef __attribute__((ext_vector_type(16))) __bf16 v16bf;
typedef __attribute__((ext_vector_type(8)))  float  v8f;

#define DEV __device__ __forceinline__

DEV unsigned short bf16_rne(float f) {
  unsigned int u = __builtin_bit_cast(unsigned int, f);
  u = u + 0x7FFFu + ((u >> 16) & 1u);
  return (unsigned short)(u >> 16);
}
DEV unsigned int pack2_bf16(float lo, float hi) {
  return (unsigned int)bf16_rne(lo) | ((unsigned int)bf16_rne(hi) << 16);
}

union BF16Frag { unsigned int u[8]; v16bf v; };

// ---------------------------------------------------------------------------
// Unified NT-GEMM / implicit-im2col conv:
//   out[n][m] = sum_{t,c} wgt[m][t*Cin+c] * act[pix(n,t)][c]
// act: bf16 NHWC [H*W][Cin]; wgt: bf16 [Cout][taps*Cin]; out: f32 NHWC.
// One wave -> 4 consecutive 16-pixel tiles x one 16-channel tile; the weight
// A-fragment is loaded once per (tap, chunk) and reused for 4 WMMAs.
// The tap loop is uniform (scalar dy/dx), so no per-lane integer division.
// ---------------------------------------------------------------------------
__global__ void __launch_bounds__(32)
conv_wmma_kernel(const unsigned short* __restrict__ act,
                 const unsigned short* __restrict__ wgt,
                 float* __restrict__ out,
                 const float* __restrict__ residual,
                 unsigned short* __restrict__ out_bf,
                 int H, int W, int Cin, int Cout, int taps)
{
  const int lane = threadIdx.x & 31;
  const int half = lane >> 4;
  const int lid  = lane & 15;
  const int K    = taps * Cin;
  const int m    = blockIdx.y * 16 + lid;   // output channel (A row)
  const int nb   = blockIdx.x * 64;         // 4 pixel tiles (B columns)

  int py[4], px[4];
#pragma unroll
  for (int it = 0; it < 4; ++it) {
    const int n = nb + it * 16 + lid;
    py[it] = n / W;
    px[it] = n - py[it] * W;
  }

  v8f acc[4] = {};
  const unsigned short* wrow = wgt + (size_t)m * K;

  for (int t = 0; t < taps; ++t) {
    const int dy = (taps == 1) ? 0 : (t / 3 - 1);   // uniform
    const int dx = (taps == 1) ? 0 : (t % 3 - 1);   // uniform
    const unsigned short* wtap = wrow + t * Cin;
    __builtin_prefetch(wtap, 0, 0);

    // per-lane source base offset for this tap (-1 => zero pad)
    int base[4];
#pragma unroll
    for (int it = 0; it < 4; ++it) {
      const int yy = py[it] + dy, xx = px[it] + dx;
      base[it] = (yy >= 0 && yy < H && xx >= 0 && xx < W)
                   ? (yy * W + xx) * Cin : -1;
    }

    for (int cb = 0; cb < Cin; cb += 32) {
      BF16Frag a;
      const bool full = (cb + 32) <= Cin;   // uniform branch
      if (full) {
#pragma unroll
        for (int j = 0; j < 8; ++j) {
          const int ka = cb + ((j < 4) ? (2 * j + 8 * half)
                                       : (16 + 2 * (j - 4) + 8 * half));
          a.u[j] = *(const unsigned int*)(wtap + ka);
        }
#pragma unroll
        for (int it = 0; it < 4; ++it) {
          BF16Frag b;
#pragma unroll
          for (int j = 0; j < 8; ++j) {
            const int kc = cb + ((j < 4) ? (2 * j + 8 * half) : (16 + 2 * (j - 4) + 8 * half));
            b.u[j] = (base[it] >= 0)
                       ? *(const unsigned int*)(act + base[it] + kc) : 0u;
          }
          acc[it] = __builtin_amdgcn_wmma_f32_16x16x32_bf16(
              false, a.v, false, b.v, (short)0, acc[it], false, false);
        }
      } else {                               // Cin % 32 == 16 tail: zero-fill
#pragma unroll
        for (int j = 0; j < 8; ++j) {
          const int ka = cb + ((j < 4) ? (2 * j + 8 * half)
                                       : (16 + 2 * (j - 4) + 8 * half));
          a.u[j] = (ka < Cin) ? *(const unsigned int*)(wtap + ka) : 0u;
        }
#pragma unroll
        for (int it = 0; it < 4; ++it) {
          BF16Frag b;
#pragma unroll
          for (int j = 0; j < 8; ++j) {
            const int kc = cb + ((j < 4) ? (2 * j + 8 * half) : (16 + 2 * (j - 4) + 8 * half));
            b.u[j] = (base[it] >= 0 && kc < Cin)
                       ? *(const unsigned int*)(act + base[it] + kc) : 0u;
          }
          acc[it] = __builtin_amdgcn_wmma_f32_16x16x32_bf16(
              false, a.v, false, b.v, (short)0, acc[it], false, false);
        }
      }
    }
  }

  const int mbase = blockIdx.y * 16 + 8 * half;
#pragma unroll
  for (int it = 0; it < 4; ++it) {
    const int n = nb + it * 16 + lid;
    float* op = out + (size_t)n * Cout + mbase;
    const float* rp = residual ? residual + (size_t)n * Cout + mbase : nullptr;
    unsigned short* bp = out_bf ? out_bf + (size_t)n * Cout + mbase : nullptr;
#pragma unroll
    for (int r = 0; r < 8; ++r) {
      float v = acc[it][r];
      if (rp) v += rp[r];
      op[r] = v;
      if (bp) bp[r] = bf16_rne(v);
    }
  }
}

// ---------------------------------------------------------------------------
// Flash attention: one wave handles 16 queries of one head. Online softmax;
// QK^T and PV both via WMMA. qkv: f32 [N][3C]; out: f32/bf16 [N][C].
// d = C/heads in {16,32}; d=16 K-dims are zero padded.
// ---------------------------------------------------------------------------
__global__ void __launch_bounds__(32)
flash_attn_kernel(const float* __restrict__ qkv,
                  float* __restrict__ outf,
                  unsigned short* __restrict__ outbf,
                  int N, int C, int heads)
{
  const int d    = C / heads;
  const int head = blockIdx.y;
  const int qb   = blockIdx.x * 16;
  const int lane = threadIdx.x & 31;
  const int half = lane >> 4;
  const int lid  = lane & 15;
  const float scale = rsqrtf((float)d);
  const int C3 = 3 * C;
  const int qo = head * d;
  const int ko = C + head * d;
  const int vo = 2 * C + head * d;

  __shared__ unsigned short pbuf[16][32];   // P tile, C-layout -> A-layout hop

  // Q A-fragment (scale folded in)
  BF16Frag qa;
#pragma unroll
  for (int j = 0; j < 8; ++j) {
    const int k0 = (j < 4) ? (2 * j + 8 * half) : (16 + 2 * (j - 4) + 8 * half);
    unsigned int v = 0u;
    if (k0 < d) {
      const float* p = qkv + (size_t)(qb + lid) * C3 + qo + k0;
      v = pack2_bf16(p[0] * scale, p[1] * scale);
    }
    qa.u[j] = v;
  }

  v8f O0 = {}, O1 = {};
  float mi[8], li[8];
#pragma unroll
  for (int r = 0; r < 8; ++r) { mi[r] = -3.0e38f; li[r] = 0.f; }

  for (int kb = 0; kb < N; kb += 32) {
    // ---- scores for 32 keys (two WMMAs) ----
    v8f S[2];
#pragma unroll
    for (int sb = 0; sb < 2; ++sb) {
      BF16Frag kf;
#pragma unroll
      for (int j = 0; j < 8; ++j) {
        const int kd = (j < 4) ? (2 * j + 8 * half) : (16 + 2 * (j - 4) + 8 * half);     // reduction dim = d-channel
        unsigned int v = 0u;
        if (kd < d) {
          const float* p = qkv + (size_t)(kb + sb * 16 + lid) * C3 + ko + kd;
          v = pack2_bf16(p[0], p[1]);
        }
        kf.u[j] = v;
      }
      v8f z = {};
      S[sb] = __builtin_amdgcn_wmma_f32_16x16x32_bf16(false, qa.v, false, kf.v,
                                                      (short)0, z, false, false);
    }
    // ---- online softmax update (row = r + 8*half, cols across 16 lanes) ----
#pragma unroll
    for (int r = 0; r < 8; ++r) {
      float mx = fmaxf(S[0][r], S[1][r]);
      mx = fmaxf(mx, __shfl_xor(mx, 1, 32));
      mx = fmaxf(mx, __shfl_xor(mx, 2, 32));
      mx = fmaxf(mx, __shfl_xor(mx, 4, 32));
      mx = fmaxf(mx, __shfl_xor(mx, 8, 32));
      const float mn = fmaxf(mi[r], mx);
      const float p0 = __expf(S[0][r] - mn);
      const float p1 = __expf(S[1][r] - mn);
      S[0][r] = p0; S[1][r] = p1;
      float rs = p0 + p1;
      rs += __shfl_xor(rs, 1, 32);
      rs += __shfl_xor(rs, 2, 32);
      rs += __shfl_xor(rs, 4, 32);
      rs += __shfl_xor(rs, 8, 32);
      const float al = __expf(mi[r] - mn);
      li[r] = li[r] * al + rs;
      O0[r] *= al;
      O1[r] *= al;
      mi[r] = mn;
    }
    __syncthreads();
#pragma unroll
    for (int r = 0; r < 8; ++r) {
      const int row = r + 8 * half;
      pbuf[row][lid]      = bf16_rne(S[0][r]);
      pbuf[row][lid + 16] = bf16_rne(S[1][r]);
    }
    __syncthreads();
    // P as A-fragment (16 queries x 32 keys)
    BF16Frag pa;
#pragma unroll
    for (int j = 0; j < 8; ++j) {
      const int k0 = (j < 4) ? (2 * j + 8 * half) : (16 + 2 * (j - 4) + 8 * half);
      pa.u[j] = *(const unsigned int*)&pbuf[lid][k0];
    }
    // ---- O += P x V ----
    const int nvb = (d + 15) >> 4;
    for (int vb = 0; vb < nvb; ++vb) {
      BF16Frag vf;
#pragma unroll
      for (int j = 0; j < 8; ++j) {
        const int kd = (j < 4) ? (2 * j + 8 * half) : (16 + 2 * (j - 4) + 8 * half);     // key index within block
        const float* p = qkv + (size_t)(kb + kd) * C3 + vo + vb * 16 + lid;
        vf.u[j] = pack2_bf16(p[0], p[C3]);    // V[key][dc], V[key+1][dc]
      }
      if (vb == 0)
        O0 = __builtin_amdgcn_wmma_f32_16x16x32_bf16(false, pa.v, false, vf.v,
                                                     (short)0, O0, false, false);
      else
        O1 = __builtin_amdgcn_wmma_f32_16x16x32_bf16(false, pa.v, false, vf.v,
                                                     (short)0, O1, false, false);
    }
  }

#pragma unroll
  for (int r = 0; r < 8; ++r) {
    const int q = qb + r + 8 * half;
    const float inv = 1.0f / li[r];
    const size_t base = (size_t)q * C + head * d + lid;
    const float v0 = O0[r] * inv;
    outf[base]  = v0;
    outbf[base] = bf16_rne(v0);
    if (d == 32) {
      const float v1 = O1[r] * inv;
      outf[base + 16]  = v1;
      outbf[base + 16] = bf16_rne(v1);
    }
  }
}

// ---------------------------------------------------------------------------
// Pointwise / reduction helpers
// ---------------------------------------------------------------------------
__global__ void cvt_bf16_kernel(const float* __restrict__ s,
                                unsigned short* __restrict__ d, int n) {
  const int i = blockIdx.x * blockDim.x + threadIdx.x;
  if (i < n) d[i] = bf16_rne(s[i]);
}

// OIHW -> [O][tap*Cin + c] bf16
__global__ void reorder_w3_kernel(const float* __restrict__ w,
                                  unsigned short* __restrict__ o, int O, int Cin) {
  const int i = blockIdx.x * blockDim.x + threadIdx.x;
  const int tot = O * Cin * 9;
  if (i >= tot) return;
  const int oc  = i / (Cin * 9);
  const int rem = i - oc * (Cin * 9);
  const int t   = rem / Cin;
  const int c   = rem - t * Cin;
  o[(size_t)oc * (9 * Cin) + t * Cin + c] =
      bf16_rne(w[((size_t)oc * Cin + c) * 9 + t]);
}

__global__ void nchw_to_nhwc_bf16_kernel(const float* __restrict__ x,
                                         unsigned short* __restrict__ o,
                                         int C, int HW) {
  const int i = blockIdx.x * blockDim.x + threadIdx.x;
  if (i >= C * HW) return;
  const int c = i / HW;
  const int p = i - c * HW;
  o[(size_t)p * C + c] = bf16_rne(x[i]);
}

__global__ void bn_stats_kernel(const float* __restrict__ x,
                                float* __restrict__ stats, int N, int C) {
  const int c = blockIdx.x;
  const int tid = threadIdx.x;
  float s = 0.f, s2 = 0.f;
  for (int n = tid; n < N; n += blockDim.x) {
    const float v = x[(size_t)n * C + c];
    s += v; s2 += v * v;
  }
  __shared__ float sh[2][256];
  sh[0][tid] = s; sh[1][tid] = s2;
  __syncthreads();
  for (int o = 128; o > 0; o >>= 1) {
    if (tid < o) { sh[0][tid] += sh[0][tid + o]; sh[1][tid] += sh[1][tid + o]; }
    __syncthreads();
  }
  if (tid == 0) {
    const float m = sh[0][0] / N;
    stats[c]     = m;
    stats[C + c] = sh[1][0] / N - m * m;
  }
}

__global__ void bn_relu_kernel(const float* __restrict__ x,
                               const float* __restrict__ stats,
                               const float* __restrict__ g,
                               const float* __restrict__ b,
                               float* __restrict__ of,
                               unsigned short* __restrict__ ob, int N, int C) {
  const int i = blockIdx.x * blockDim.x + threadIdx.x;
  if (i >= N * C) return;
  const int c = i % C;
  float y = (x[i] - stats[c]) * rsqrtf(stats[C + c] + 1e-5f) * g[c] + b[c];
  y = fmaxf(y, 0.f);
  of[i] = y;
  ob[i] = bf16_rne(y);
}

__global__ void maxpool2_kernel(const float* __restrict__ x,
                                unsigned short* __restrict__ o,
                                int H, int W, int C) {
  const int OW = W >> 1, OH = H >> 1;
  const int i = blockIdx.x * blockDim.x + threadIdx.x;
  if (i >= OH * OW * C) return;
  const int c = i % C;
  const int p = i / C;
  const int ox = p % OW, oy = p / OW;
  const float* b0 = x + ((size_t)(2 * oy) * W + 2 * ox) * C + c;
  const size_t rs = (size_t)W * C;
  const float m = fmaxf(fmaxf(b0[0], b0[C]), fmaxf(b0[rs], b0[rs + C]));
  o[i] = bf16_rne(m);
}

__global__ void upsample_ac_kernel(const float* __restrict__ x,
                                   float* __restrict__ of,
                                   unsigned short* __restrict__ ob,
                                   int H, int W, int C, int OH, int OW) {
  const int i = blockIdx.x * blockDim.x + threadIdx.x;
  if (i >= OH * OW * C) return;
  const int c = i % C;
  const int p = i / C;
  const int ox = p % OW, oy = p / OW;
  const float sy = (OH > 1) ? (float)oy * (float)(H - 1) / (float)(OH - 1) : 0.f;
  const float sx = (OW > 1) ? (float)ox * (float)(W - 1) / (float)(OW - 1) : 0.f;
  const int y0 = (int)sy, x0 = (int)sx;
  const int y1 = (y0 + 1 < H) ? y0 + 1 : H - 1;
  const int x1 = (x0 + 1 < W) ? x0 + 1 : W - 1;
  const float wy = sy - y0, wx = sx - x0;
  const float t00 = x[((size_t)y0 * W + x0) * C + c];
  const float t01 = x[((size_t)y0 * W + x1) * C + c];
  const float t10 = x[((size_t)y1 * W + x0) * C + c];
  const float t11 = x[((size_t)y1 * W + x1) * C + c];
  const float top = t00 * (1.f - wx) + t01 * wx;
  const float bot = t10 * (1.f - wx) + t11 * wx;
  const float v = top * (1.f - wy) + bot * wy;
  if (of) of[i] = v;
  if (ob) ob[i] = bf16_rne(v);
}

__global__ void copych_kernel(const unsigned short* __restrict__ s,
                              unsigned short* __restrict__ d,
                              int N, int Cs, int Cd, int off) {
  const int i = blockIdx.x * blockDim.x + threadIdx.x;
  if (i >= N * Cs) return;
  const int c = i % Cs;
  const int n = i / Cs;
  d[(size_t)n * Cd + off + c] = s[i];
}

__global__ void outc_kernel(const float* __restrict__ y,
                            const float* __restrict__ w,
                            const float* __restrict__ b,
                            float* __restrict__ o, int N, int C) {
  const int n = blockIdx.x * blockDim.x + threadIdx.x;
  if (n >= N) return;
  float s = b[0];
  for (int c = 0; c < C; ++c) s += w[c] * y[(size_t)n * C + c];
  o[n] = s;
}

// ---------------------------------------------------------------------------
// Host orchestration
// ---------------------------------------------------------------------------
extern "C" void kernel_launch(void* const* d_in, const int* in_sizes, int n_in,
                              void* d_out, int out_size, void* d_ws, size_t ws_size,
                              hipStream_t stream) {
  (void)in_sizes; (void)n_in; (void)out_size; (void)ws_size;

  int idx = 0;
  const float* X = (const float*)d_in[idx++];           // (1,512,64,64)
  struct DCp { const float *w1,*g1,*b1,*w2,*g2,*b2; };
  auto getDC = [&]() {
    DCp p;
    p.w1=(const float*)d_in[idx++]; p.g1=(const float*)d_in[idx++];
    p.b1=(const float*)d_in[idx++]; p.w2=(const float*)d_in[idx++];
    p.g2=(const float*)d_in[idx++]; p.b2=(const float*)d_in[idx++];
    return p;
  };
  struct ATp { const float *qkv, *out; };
  auto getAT = [&]() {
    ATp a;
    a.qkv=(const float*)d_in[idx++]; a.out=(const float*)d_in[idx++];
    return a;
  };
  DCp inc = getDC(), dn1 = getDC(), dn2 = getDC(), dn3 = getDC();
  ATp at1 = getAT(), at2 = getAT(), at3 = getAT(), at4 = getAT();
  const float* ap1w = (const float*)d_in[idx++]; DCp up1 = getDC();
  const float* ap2w = (const float*)d_in[idx++]; DCp up2 = getDC();
  const float* ap3w = (const float*)d_in[idx++]; DCp up3 = getDC();
  const float* ocw = (const float*)d_in[idx++];
  const float* ocb = (const float*)d_in[idx++];

  // --- workspace arena ---
  size_t off = 0;
  auto AA = [&](size_t bytes) {
    void* p = (char*)d_ws + off;
    off += (bytes + 255) & ~(size_t)255;
    return p;
  };
  auto AF = [&](size_t n) { return (float*)AA(n * 4); };
  auto AB = [&](size_t n) { return (unsigned short*)AA(n * 2); };

  const int N1 = 4096, N2 = 1024, N3 = 256, N4 = 64;

  // weights (bf16, rebuilt every launch; deterministic)
  unsigned short *w_inc1=AB(64*4608), *w_inc2=AB(64*576);
  unsigned short *w_d1a=AB(128*576),  *w_d1b=AB(128*1152);
  unsigned short *w_d2a=AB(256*1152), *w_d2b=AB(256*2304);
  unsigned short *w_d3a=AB(256*2304), *w_d3b=AB(256*2304);
  unsigned short *w_u1a=AB(288*5184), *w_u1b=AB(128*2592);
  unsigned short *w_u2a=AB(144*2592), *w_u2b=AB(64*1296);
  unsigned short *w_u3a=AB(80*1440),  *w_u3b=AB(64*720);
  unsigned short *w_q1=AB(192*64),  *w_o1=AB(64*64);
  unsigned short *w_q2=AB(384*128), *w_o2=AB(128*128);
  unsigned short *w_q3=AB(768*256), *w_o3=AB(256*256);
  unsigned short *w_q4=AB(768*256), *w_o4=AB(256*256);
  unsigned short *w_p1=AB(64*256), *w_p2=AB(32*128), *w_p3=AB(32*64);
  float* stats = AF(1024);

  // activations (NHWC)
  unsigned short* xbf  = AB((size_t)N1*512);
  float* t1 = AF((size_t)N1*64);  unsigned short* h1bf = AB((size_t)N1*64);
  float* x1f = AF((size_t)N1*64); unsigned short* x1bf = AB((size_t)N1*64);
  float* qk1 = AF((size_t)N1*192);
  float* ao1f = AF((size_t)N1*64); unsigned short* ao1bf = AB((size_t)N1*64);
  float* x1af = AF((size_t)N1*64); unsigned short* x1abf = AB((size_t)N1*64);
  unsigned short* p1bf = AB((size_t)N2*64);
  float* t2 = AF((size_t)N2*128);  unsigned short* h2bf = AB((size_t)N2*128);
  float* x2f = AF((size_t)N2*128); unsigned short* x2bf = AB((size_t)N2*128);
  float* qk2 = AF((size_t)N2*384);
  float* ao2f = AF((size_t)N2*128); unsigned short* ao2bf = AB((size_t)N2*128);
  float* x2af = AF((size_t)N2*128); unsigned short* x2abf = AB((size_t)N2*128);
  unsigned short* p2bf = AB((size_t)N3*128);
  float* t3 = AF((size_t)N3*256);  unsigned short* h3bf = AB((size_t)N3*256);
  float* x3f = AF((size_t)N3*256); unsigned short* x3bf = AB((size_t)N3*256);
  float* qk3 = AF((size_t)N3*768);
  float* ao3f = AF((size_t)N3*256); unsigned short* ao3bf = AB((size_t)N3*256);
  float* x3af = AF((size_t)N3*256); unsigned short* x3abf = AB((size_t)N3*256);
  unsigned short* p3bf = AB((size_t)N4*256);
  float* t4 = AF((size_t)N4*256);  unsigned short* h4bf = AB((size_t)N4*256);
  float* x4f = AF((size_t)N4*256); unsigned short* x4bf = AB((size_t)N4*256);
  float* qk4 = AF((size_t)N4*768);
  float* ao4f = AF((size_t)N4*256); unsigned short* ao4bf = AB((size_t)N4*256);
  float* x4af = AF((size_t)N4*256); unsigned short* x4abf = AB((size_t)N4*256);
  float* u1f = AF((size_t)N3*256); unsigned short* u1bf = AB((size_t)N3*256);
  float* apo1f = AF((size_t)N3*64); unsigned short* apo1bf = AB((size_t)N3*64);
  unsigned short* cat1 = AB((size_t)N3*576);
  float* tu1 = AF((size_t)N3*288); unsigned short* hu1bf = AB((size_t)N3*288);
  float* y1f = AF((size_t)N3*128); unsigned short* y1bf = AB((size_t)N3*128);
  float* u2f = AF((size_t)N2*128); unsigned short* u2bf = AB((size_t)N2*128);
  float* apo2f = AF((size_t)N2*32); unsigned short* apo2bf = AB((size_t)N2*32);
  unsigned short* cat2 = AB((size_t)N2*288);
  float* tu2 = AF((size_t)N2*144); unsigned short* hu2bf = AB((size_t)N2*144);
  float* y2f = AF((size_t)N2*64);  unsigned short* y2bf = AB((size_t)N2*64);
  float* u3f = AF((size_t)N1*64);  unsigned short* u3bf = AB((size_t)N1*64);
  float* apo3f = AF((size_t)N1*32); unsigned short* apo3bf = AB((size_t)N1*32);
  unsigned short* cat3 = AB((size_t)N1*160);
  float* tu3 = AF((size_t)N1*80);  unsigned short* hu3bf = AB((size_t)N1*80);
  float* y3f = AF((size_t)N1*64);  unsigned short* y3bf = AB((size_t)N1*64);
  float* logits = AF((size_t)N1);

  // --- launch helpers ---
  auto reorder3 = [&](const float* w, unsigned short* dst, int O, int Cin) {
    const int n = O * Cin * 9;
    reorder_w3_kernel<<<(n + 255) / 256, 256, 0, stream>>>(w, dst, O, Cin);
  };
  auto cvt = [&](const float* s, unsigned short* dst, int n) {
    cvt_bf16_kernel<<<(n + 255) / 256, 256, 0, stream>>>(s, dst, n);
  };
  auto conv = [&](const unsigned short* act, const unsigned short* w, float* o,
                  const float* res, unsigned short* obf,
                  int H, int W, int Cin, int Cout, int taps) {
    dim3 g((unsigned)((H * W) / 64), (unsigned)(Cout / 16));
    conv_wmma_kernel<<<g, 32, 0, stream>>>(act, w, o, res, obf,
                                           H, W, Cin, Cout, taps);
  };
  auto bn = [&](float* xio, const float* g_, const float* b_,
                unsigned short* ob, int N, int C) {
    bn_stats_kernel<<<C, 256, 0, stream>>>(xio, stats, N, C);
    const int n = N * C;
    bn_relu_kernel<<<(n + 255) / 256, 256, 0, stream>>>(xio, stats, g_, b_,
                                                        xio, ob, N, C);
  };
  auto dconv = [&](const unsigned short* inbf, int H, int W,
                   int Cin, int Cmid, int Cout, const DCp& p,
                   const unsigned short* w1, const unsigned short* w2,
                   float* t, unsigned short* hbf, float* of, unsigned short* obf) {
    conv(inbf, w1, t, nullptr, nullptr, H, W, Cin, Cmid, 9);
    bn(t, p.g1, p.b1, hbf, H * W, Cmid);
    conv(hbf, w2, of, nullptr, nullptr, H, W, Cmid, Cout, 9);
    bn(of, p.g2, p.b2, obf, H * W, Cout);
  };
  auto attn = [&](const float* xf, const unsigned short* xb, int H, int W,
                  int C, int heads, const unsigned short* wq,
                  const unsigned short* wo, float* qkvbuf,
                  float* aof, unsigned short* aobf,
                  float* of, unsigned short* obf) {
    const int N = H * W;
    conv(xb, wq, qkvbuf, nullptr, nullptr, H, W, C, 3 * C, 1);
    dim3 g((unsigned)(N / 16), (unsigned)heads);
    flash_attn_kernel<<<g, 32, 0, stream>>>(qkvbuf, aof, aobf, N, C, heads);
    conv(aobf, wo, of, xf, obf, H, W, C, C, 1);   // out-proj + residual
  };
  auto pool = [&](const float* xf, unsigned short* obf, int H, int W, int C) {
    const int n = (H / 2) * (W / 2) * C;
    maxpool2_kernel<<<(n + 255) / 256, 256, 0, stream>>>(xf, obf, H, W, C);
  };
  auto upsample = [&](const float* xf, float* of, unsigned short* ob,
                      int H, int W, int C, int OH, int OW) {
    const int n = OH * OW * C;
    upsample_ac_kernel<<<(n + 255) / 256, 256, 0, stream>>>(xf, of, ob,
                                                            H, W, C, OH, OW);
  };
  auto copych = [&](const unsigned short* s, unsigned short* dmem,
                    int N, int Cs, int Cd, int o) {
    const int n = N * Cs;
    copych_kernel<<<(n + 255) / 256, 256, 0, stream>>>(s, dmem, N, Cs, Cd, o);
  };

  // --- weight conversion ---
  reorder3(inc.w1, w_inc1, 64, 512); reorder3(inc.w2, w_inc2, 64, 64);
  reorder3(dn1.w1, w_d1a, 128, 64);  reorder3(dn1.w2, w_d1b, 128, 128);
  reorder3(dn2.w1, w_d2a, 256, 128); reorder3(dn2.w2, w_d2b, 256, 256);
  reorder3(dn3.w1, w_d3a, 256, 256); reorder3(dn3.w2, w_d3b, 256, 256);
  reorder3(up1.w1, w_u1a, 288, 576); reorder3(up1.w2, w_u1b, 128, 288);
  reorder3(up2.w1, w_u2a, 144, 288); reorder3(up2.w2, w_u2b, 64, 144);
  reorder3(up3.w1, w_u3a, 80, 160);  reorder3(up3.w2, w_u3b, 64, 80);
  cvt(at1.qkv, w_q1, 192*64);  cvt(at1.out, w_o1, 64*64);
  cvt(at2.qkv, w_q2, 384*128); cvt(at2.out, w_o2, 128*128);
  cvt(at3.qkv, w_q3, 768*256); cvt(at3.out, w_o3, 256*256);
  cvt(at4.qkv, w_q4, 768*256); cvt(at4.out, w_o4, 256*256);
  cvt(ap1w, w_p1, 64*256); cvt(ap2w, w_p2, 32*128); cvt(ap3w, w_p3, 32*64);

  // --- encoder ---
  { const int n = 512 * N1;
    nchw_to_nhwc_bf16_kernel<<<(n + 255) / 256, 256, 0, stream>>>(X, xbf, 512, N1); }

  dconv(xbf, 64, 64, 512, 64, 64, inc, w_inc1, w_inc2, t1, h1bf, x1f, x1bf);
  attn(x1f, x1bf, 64, 64, 64, 4, w_q1, w_o1, qk1, ao1f, ao1bf, x1af, x1abf);
  pool(x1f, p1bf, 64, 64, 64);
  dconv(p1bf, 32, 32, 64, 128, 128, dn1, w_d1a, w_d1b, t2, h2bf, x2f, x2bf);
  attn(x2f, x2bf, 32, 32, 128, 4, w_q2, w_o2, qk2, ao2f, ao2bf, x2af, x2abf);
  pool(x2f, p2bf, 32, 32, 128);
  dconv(p2bf, 16, 16, 128, 256, 256, dn2, w_d2a, w_d2b, t3, h3bf, x3f, x3bf);
  attn(x3f, x3bf, 16, 16, 256, 8, w_q3, w_o3, qk3, ao3f, ao3bf, x3af, x3abf);
  pool(x3f, p3bf, 16, 16, 256);
  dconv(p3bf, 8, 8, 256, 256, 256, dn3, w_d3a, w_d3b, t4, h4bf, x4f, x4bf);
  attn(x4f, x4bf, 8, 8, 256, 8, w_q4, w_o4, qk4, ao4f, ao4bf, x4af, x4abf);

  // --- decoder ---
  upsample(x4af, u1f, u1bf, 8, 8, 256, 16, 16);
  conv(x3abf, w_p1, apo1f, nullptr, apo1bf, 16, 16, 256, 64, 1);
  copych(u1bf, cat1, N3, 256, 576, 0);
  copych(x3bf, cat1, N3, 256, 576, 256);
  copych(apo1bf, cat1, N3, 64, 576, 512);
  dconv(cat1, 16, 16, 576, 288, 128, up1, w_u1a, w_u1b, tu1, hu1bf, y1f, y1bf);

  upsample(y1f, u2f, u2bf, 16, 16, 128, 32, 32);
  conv(x2abf, w_p2, apo2f, nullptr, apo2bf, 32, 32, 128, 32, 1);
  copych(u2bf, cat2, N2, 128, 288, 0);
  copych(x2bf, cat2, N2, 128, 288, 128);
  copych(apo2bf, cat2, N2, 32, 288, 256);
  dconv(cat2, 32, 32, 288, 144, 64, up2, w_u2a, w_u2b, tu2, hu2bf, y2f, y2bf);

  upsample(y2f, u3f, u3bf, 32, 32, 64, 64, 64);
  conv(x1abf, w_p3, apo3f, nullptr, apo3bf, 64, 64, 64, 32, 1);
  copych(u3bf, cat3, N1, 64, 160, 0);
  copych(x1bf, cat3, N1, 64, 160, 64);
  copych(apo3bf, cat3, N1, 32, 160, 96);
  dconv(cat3, 64, 64, 160, 80, 64, up3, w_u3a, w_u3b, tu3, hu3bf, y3f, y3bf);

  outc_kernel<<<(N1 + 255) / 256, 256, 0, stream>>>(y3f, ocw, ocb, logits, N1, 64);
  { const int n = 256 * 256;
    upsample_ac_kernel<<<(n + 255) / 256, 256, 0, stream>>>(
        logits, (float*)d_out, nullptr, 64, 64, 1, 256, 256); }
}
